// MeshSegNet_39444979647057
// MI455X (gfx1250) — matrix operand order repaired and checked by reading, not compiled, not run on hardware
//
#include <hip/hip_runtime.h>
#include <hip/hip_bf16.h>

#define NPAD 6016
#define NREAL 6000

typedef __attribute__((ext_vector_type(16))) _Float16 v16h;
typedef __attribute__((ext_vector_type(8)))  float    v8f;
typedef __attribute__((ext_vector_type(4)))  unsigned int uint4v;
typedef __attribute__((ext_vector_type(4)))  int i4;

// ---------------------------------------------------------------------------
// CDNA5 async global->LDS copy (16B per lane), with plain-copy fallback.
// Builtin signature (from hipcc diagnostic): non-const v4i32* args.
// ---------------------------------------------------------------------------
#if __has_builtin(__builtin_amdgcn_global_load_async_to_lds_b128)
#define HAVE_ASYNC 1
#else
#define HAVE_ASYNC 0
#endif

__device__ __forceinline__ void cp16(const _Float16* g, _Float16* l) {
#if HAVE_ASYNC
  void* gv = (void*)g;  // strip const
  void* lv = (void*)l;
  __builtin_amdgcn_global_load_async_to_lds_b128(
      (__attribute__((address_space(1))) i4*)gv,
      (__attribute__((address_space(3))) i4*)lv, 0, 0);
#else
  *(uint4v*)l = *(const uint4v*)g;
#endif
}

__device__ __forceinline__ void async_wait0() {
#if HAVE_ASYNC
#if __has_builtin(__builtin_amdgcn_s_wait_asynccnt)
  __builtin_amdgcn_s_wait_asynccnt(0);
#else
  asm volatile("s_wait_asynccnt 0x0" ::: "memory");
#endif
#endif
}

__device__ __forceinline__ v8f wmma16(v16h a, v16h b, v8f c) {
  return __builtin_amdgcn_wmma_f32_16x16x32_f16(false, a, false, b, (short)0, c,
                                                false, false);
}

// A-fragment (16x32 f16): lane L holds row m = L%16; VGPR v holds K pairs:
// v<4 -> K = 2v,2v+1 ; v>=4 -> K = 2v+8,2v+9 ; +8 for lanes >= 16.
__device__ __forceinline__ void load_a_frag(const _Float16* lds, int stride,
                                            int mbase, int lane, v16h& a) {
  const _Float16* rp = lds + (size_t)(mbase + (lane & 15)) * stride;
  int g8 = (lane >> 4) * 8;
#pragma unroll
  for (int v = 0; v < 8; ++v) {
    int kb = ((v < 4) ? (2 * v) : (2 * v + 8)) + g8;
    a[2 * v]     = rp[kb];
    a[2 * v + 1] = rp[kb + 1];
  }
}

// ---------------------------------------------------------------------------
// Generic 1x1 Conv (GEMM: out[o,n] = relu(bn(sum_i W[o,i]*x[i,n] + bias)))
// f16 activations, fp32 weights/accum. Block = 128 thr (4 waves).
// Tile 32(o) x 64(n); K-chunks of 32. Inputs are zero-padded to 32-row
// multiples and NPAD columns, so B staging is a guard-free async byte copy.
// ---------------------------------------------------------------------------
template <typename OT>
__global__ __launch_bounds__(128) void conv_wmma_kernel(
    const _Float16* __restrict__ x, long xbs,
    const float* __restrict__ W, const float* __restrict__ bias,
    const float* __restrict__ bng, const float* __restrict__ bnbe,
    const float* __restrict__ bnm, const float* __restrict__ bnv,
    int relu, int O, int I,
    OT* __restrict__ out, long obs, int ors, int row0, int nvout) {
  __shared__ _Float16 As[32][40];   // 80B rows (16B aligned)
  __shared__ _Float16 Bs[32][72];   // 144B rows (16B aligned)
  __shared__ float scale_s[32];
  __shared__ float shift_s[32];

  const int tid = threadIdx.x;
  const int b  = blockIdx.z;
  const int n0 = blockIdx.x * 64;
  const int o0 = blockIdx.y * 32;

  if (tid < 32) {
    int o = o0 + tid;
    float sc = 1.0f, sh = 0.0f;
    if (o < O) {
      float bb = bias ? bias[o] : 0.0f;
      if (bng) {
        sc = bng[o] * rsqrtf(bnv[o] + 1e-5f);
        sh = bnbe[o] + sc * (bb - bnm[o]);
      } else {
        sh = bb;
      }
    }
    scale_s[tid] = sc;
    shift_s[tid] = sh;
  }

  const int wave = tid >> 5, lane = tid & 31;
  const int osub  = wave & 1;
  const int nsub0 = (wave >> 1) * 2;

  v8f accs[2];
  {
    v8f z = {0.f, 0.f, 0.f, 0.f, 0.f, 0.f, 0.f, 0.f};
    accs[0] = z; accs[1] = z;
  }

  const _Float16* xb = x + (long)b * xbs;
  const int kchunks = (I + 31) / 32;
  const int brow = tid >> 2;            // 0..31
  const int bseg = (tid & 3) * 16;      // 0,16,32,48

  for (int kc = 0; kc < kchunks; ++kc) {
    const int i0 = kc * 32;
    __syncthreads();
    // Stage W tile 32x32 (fp32 -> f16, guarded)
#pragma unroll
    for (int j = 0; j < 8; ++j) {
      int idx = tid * 8 + j;
      int r = idx >> 5, c = idx & 31;
      int o = o0 + r, i = i0 + c;
      As[r][c] = (_Float16)((o < O && i < I) ? W[(long)o * I + i] : 0.0f);
    }
    // Stage X tile 32(i) x 64(n): pure byte copy (async on CDNA5)
    {
      const _Float16* src = xb + (long)(i0 + brow) * NPAD + n0 + bseg;
      cp16(src, &Bs[brow][bseg]);
      cp16(src + 8, &Bs[brow][bseg + 8]);
      if (kc + 1 < kchunks) __builtin_prefetch(src + 32 * NPAD, 0, 1);
    }
    async_wait0();
    __syncthreads();

    v16h a;
    load_a_frag(&As[0][0], 40, osub * 16, lane, a);
#pragma unroll
    for (int t = 0; t < 2; ++t) {
      v16h bf;
      {
        int n = (nsub0 + t) * 16 + (lane & 15);
        int k0 = (lane >> 4) * 8;
#pragma unroll
        for (int e = 0; e < 16; ++e) bf[e] = Bs[k0 + (e < 8 ? e : e + 8)][n];
      }
      accs[t] = wmma16(a, bf, accs[t]);
    }
  }

  // Epilogue: C layout: elem r, lane L -> M = r + 8*(L>>4), N = L&15
  const int gg = lane >> 4, col = lane & 15;
#pragma unroll
  for (int t = 0; t < 2; ++t) {
    int n = n0 + (nsub0 + t) * 16 + col;
#pragma unroll
    for (int r = 0; r < 8; ++r) {
      int ol = osub * 16 + r + 8 * gg;
      int o = o0 + ol;
      float yv = accs[t][r] * scale_s[ol] + shift_s[ol];
      if (relu) yv = fmaxf(yv, 0.0f);
      if (o < O && n < nvout)
        out[(long)b * obs + (long)(row0 + o) * ors + n] = (OT)yv;
    }
  }
}

// ---------------------------------------------------------------------------
// Fused adjacency GEMM: out[b,k,m] = inv[m] * sum_n (d2(m,n)<thr2) * X[b,k,n]
// A (= 0/1 S matrix) is computed straight into fragment registers from
// coords (each wave owns its 16 m-rows exclusively). B-tile is staged
// k-major so each fragment load is two 16-byte LDS runs.
// Tile: 64(m) x KSUBS*16(k); K-loop over n in 32-chunks. DUAL computes both
// thresholds (0.1 / 0.2) sharing distances and X staging.
// ---------------------------------------------------------------------------
template <int DUAL, int KSUBS>
__global__ __launch_bounds__(128) void adj_gemm_kernel(
    const float* __restrict__ coords,   // [B][NPAD][3]
    const float* __restrict__ inv_s, const float* __restrict__ inv_l,
    const _Float16* __restrict__ X, long xbs,
    _Float16* __restrict__ out_s, long osbs,
    _Float16* __restrict__ out_l, long olbs) {
  constexpr int KT = KSUBS * 16;
  __shared__ float cn[32 * 3];
  __shared__ _Float16 Xl[KT][48];     // 96B rows (16B aligned)

  const int tid = threadIdx.x;
  const int b  = blockIdx.z;
  const int m0 = blockIdx.x * 64;
  const int k0 = blockIdx.y * KT;
  const int wave = tid >> 5, lane = tid & 31;

  const int mrow = m0 + wave * 16 + (lane & 15);
  const float cmx = coords[((long)b * NPAD + mrow) * 3 + 0];
  const float cmy = coords[((long)b * NPAD + mrow) * 3 + 1];
  const float cmz = coords[((long)b * NPAD + mrow) * 3 + 2];
  const int g8 = (lane >> 4) * 8;

  v8f accS[KSUBS], accL[KSUBS];
  {
    v8f z = {0.f, 0.f, 0.f, 0.f, 0.f, 0.f, 0.f, 0.f};
#pragma unroll
    for (int t = 0; t < KSUBS; ++t) { accS[t] = z; accL[t] = z; }
  }

  const _Float16* Xb = X + (long)b * xbs + (long)k0 * NPAD;

  for (int nc = 0; nc < NPAD; nc += 32) {
    __syncthreads();
    for (int idx = tid; idx < 96; idx += 128)
      cn[idx] = coords[((long)b * NPAD + nc) * 3 + idx];
    // X tile: KT rows x 32 f16, in 32B units (2x b128 per unit)
    for (int u = tid; u < KT * 2; u += 128) {
      int r = u >> 1, s = (u & 1) * 16;
      const _Float16* src = Xb + (long)r * NPAD + nc + s;
      cp16(src, &Xl[r][s]);
      cp16(src + 8, &Xl[r][s + 8]);
      if (nc + 32 < NPAD) __builtin_prefetch(src + 32, 0, 1);
    }
    async_wait0();
    __syncthreads();

    // A fragments straight from registers (distance threshold, exact 0/1)
    v16h aS, aL;
#pragma unroll
    for (int v = 0; v < 8; ++v) {
      int kb = ((v < 4) ? (2 * v) : (2 * v + 8)) + g8;
#pragma unroll
      for (int j = 0; j < 2; ++j) {
        int nl = kb + j;
        float dx = cmx - cn[nl * 3 + 0];
        float dy = cmy - cn[nl * 3 + 1];
        float dz = cmz - cn[nl * 3 + 2];
        float d2 = dx * dx + dy * dy + dz * dz;
        bool valid = (nc + nl) < NREAL;
        aS[2 * v + j] = (_Float16)((valid && d2 < 0.01f) ? 1.0f : 0.0f);
        if (DUAL)
          aL[2 * v + j] = (_Float16)((valid && d2 < 0.04f) ? 1.0f : 0.0f);
      }
    }
#pragma unroll
    for (int ks = 0; ks < KSUBS; ++ks) {
      v16h bf;
      {
        const _Float16* p = &Xl[ks * 16 + (lane & 15)][(lane >> 4) * 8];
#pragma unroll
        for (int e = 0; e < 16; ++e) bf[e] = p[e < 8 ? e : e + 8];   // two 16-byte runs
      }
      accS[ks] = wmma16(aS, bf, accS[ks]);
      if (DUAL) accL[ks] = wmma16(aL, bf, accL[ks]);
    }
  }

  const int gg = lane >> 4, col = lane & 15;
#pragma unroll
  for (int ks = 0; ks < KSUBS; ++ks) {
    int k = k0 + ks * 16 + col;
#pragma unroll
    for (int r = 0; r < 8; ++r) {
      int m = m0 + wave * 16 + r + 8 * gg;
      float is = inv_s[(long)b * NPAD + m];
      out_s[(long)b * osbs + (long)k * NPAD + m] = (_Float16)(accS[ks][r] * is);
      if (DUAL) {
        float il = inv_l[(long)b * NPAD + m];
        out_l[(long)b * olbs + (long)k * NPAD + m] =
            (_Float16)(accL[ks][r] * il);
      }
    }
  }
}

// ---------------------------------------------------------------------------
// Small helper kernels
// ---------------------------------------------------------------------------
// fp32 input x[2,6,6000] -> f16 xh[2][32][NPAD], zero padded (rows 6..31, cols)
__global__ void convert_x_kernel(const float* __restrict__ x,
                                 _Float16* __restrict__ xh) {
  int t = blockIdx.x * blockDim.x + threadIdx.x;
  if (t >= 2 * 32 * NPAD) return;
  int b = t / (32 * NPAD);
  int rem = t % (32 * NPAD);
  int c = rem / NPAD, n = rem % NPAD;
  float v = (c < 6 && n < NREAL) ? x[(long)b * 36000 + (long)c * NREAL + n]
                                 : 0.0f;
  xh[t] = (_Float16)v;
}

__global__ void coords_copy_kernel(const float* __restrict__ x,
                                   float* __restrict__ cb) {
  int t = blockIdx.x * blockDim.x + threadIdx.x;
  if (t >= 2 * NPAD) return;
  int b = t / NPAD, n = t % NPAD;
#pragma unroll
  for (int c = 0; c < 3; ++c)
    cb[((long)b * NPAD + n) * 3 + c] =
        (n < NREAL) ? x[(long)b * 36000 + (long)c * NREAL + n] : 0.0f;
}

__global__ __launch_bounds__(128) void rowsum_kernel(
    const float* __restrict__ cb, float* __restrict__ inv_s,
    float* __restrict__ inv_l) {
  __shared__ float ch[512 * 3];
  int tid = threadIdx.x;
  int b = blockIdx.y;
  int m = blockIdx.x * 128 + tid;
  float mx = cb[((long)b * NPAD + m) * 3 + 0];
  float my = cb[((long)b * NPAD + m) * 3 + 1];
  float mz = cb[((long)b * NPAD + m) * 3 + 2];
  int cs = 0, cl = 0;
  for (int base = 0; base < NREAL; base += 512) {
    __syncthreads();
    for (int idx = tid; idx < 1536; idx += 128) {
      int n = base + idx / 3;
      ch[idx] = (n < NREAL) ? cb[((long)b * NPAD + n) * 3 + (idx % 3)] : 1.0e9f;
    }
    __syncthreads();
    for (int j = 0; j < 512; ++j) {
      float dx = mx - ch[j * 3 + 0];
      float dy = my - ch[j * 3 + 1];
      float dz = mz - ch[j * 3 + 2];
      float d2 = dx * dx + dy * dy + dz * dz;
      cs += (d2 < 0.01f) ? 1 : 0;
      cl += (d2 < 0.04f) ? 1 : 0;
    }
  }
  inv_s[(long)b * NPAD + m] = 1.0f / ((float)cs + 1e-5f);
  inv_l[(long)b * NPAD + m] = 1.0f / ((float)cl + 1e-5f);
}

__global__ void fc_kernel(const float* __restrict__ in,
                          const float* __restrict__ W,
                          const float* __restrict__ bias,
                          const float* __restrict__ g,
                          const float* __restrict__ be,
                          const float* __restrict__ m,
                          const float* __restrict__ v,
                          int relu, int add_eye, int I, int O,
                          float* __restrict__ out) {
  int t = blockIdx.x * blockDim.x + threadIdx.x;
  if (t >= 2 * O) return;
  int b = t / O, o = t % O;
  float acc = bias[o];
  const float* ip = in + (long)b * I;
  const float* wp = W + (long)o * I;
  for (int i = 0; i < I; ++i) acc += ip[i] * wp[i];
  if (g) {
    float sc = g[o] * rsqrtf(v[o] + 1e-5f);
    acc = sc * (acc - m[o]) + be[o];
  }
  if (relu) acc = fmaxf(acc, 0.0f);
  if (add_eye && (o / 64) == (o % 64)) acc += 1.0f;
  out[(long)b * O + o] = acc;
}

// x_ftm[b,k,n] = sum_c h2[b,c,n] * trans[b, c*64+k]
__global__ void xftm_kernel(const _Float16* __restrict__ h2,
                            const float* __restrict__ trans,
                            _Float16* __restrict__ out, long obs) {
  int t = blockIdx.x * blockDim.x + threadIdx.x;
  if (t >= 2 * 64 * NPAD) return;
  int b = t / (64 * NPAD);
  int rem = t % (64 * NPAD);
  int k = rem / NPAD, n = rem % NPAD;
  float acc = 0.0f;
  const _Float16* hb = h2 + (long)b * 64 * NPAD + n;
  const float* tb = trans + (long)b * 4096 + k;
  for (int c = 0; c < 64; ++c) acc += (float)hb[(long)c * NPAD] * tb[c * 64];
  out[(long)b * obs + (long)k * NPAD + n] = (_Float16)acc;
}

__global__ __launch_bounds__(256) void maxpool_kernel(
    const _Float16* __restrict__ in, long bs, int C, int nvalid,
    float* __restrict__ out) {
  __shared__ float red[256];
  int bc = blockIdx.x;
  int b = bc / C, c = bc % C;
  const _Float16* p = in + (long)b * bs + (long)c * NPAD;
  float mx = -3.4e38f;
  for (int n = threadIdx.x; n < nvalid; n += 256) mx = fmaxf(mx, (float)p[n]);
  red[threadIdx.x] = mx;
  __syncthreads();
  for (int s = 128; s > 0; s >>= 1) {
    if (threadIdx.x < s)
      red[threadIdx.x] = fmaxf(red[threadIdx.x], red[threadIdx.x + s]);
    __syncthreads();
  }
  if (threadIdx.x == 0) out[bc] = red[0];
}

__global__ void broadcast_gmax_kernel(const float* __restrict__ gmax,
                                      _Float16* __restrict__ cat) {
  int t = blockIdx.x * blockDim.x + threadIdx.x;
  if (t >= 2 * 512 * NPAD) return;
  int b = t / (512 * NPAD);
  int rem = t % (512 * NPAD);
  int c = rem / NPAD, n = rem % NPAD;
  cat[(long)b * 1600 * NPAD + (long)c * NPAD + n] = (_Float16)gmax[b * 512 + c];
}

// ---------------------------------------------------------------------------
extern "C" void kernel_launch(void* const* d_in, const int* in_sizes, int n_in,
                              void* d_out, int out_size, void* d_ws,
                              size_t ws_size, hipStream_t stream) {
  (void)in_sizes; (void)n_in; (void)out_size; (void)ws_size;
  auto F = [&](int i) { return (const float*)d_in[i]; };
  auto BN = [&](int bnb, int k) { return bnb >= 0 ? F(bnb + k) : nullptr; };

  char* base = (char*)d_ws;
  size_t off = 0;
  auto allocH = [&](size_t ne) {
    _Float16* p = (_Float16*)(base + off);
    off = (off + ne * 2 + 15) & ~(size_t)15;
    return p;
  };
  auto allocF = [&](size_t ne) {
    float* p = (float*)(base + off);
    off = (off + ne * 4 + 15) & ~(size_t)15;
    return p;
  };

  const long CH = NPAD;
  _Float16* cat    = allocH((size_t)2 * 1600 * CH);  // [gmax|x_ftm|x_mlp2|x_glm2]
  _Float16* t512a  = allocH((size_t)2 * 512 * CH);
  _Float16* t512b  = allocH((size_t)2 * 512 * CH);
  _Float16* cat384 = allocH((size_t)2 * 384 * CH);
  _Float16* t256a  = allocH((size_t)2 * 256 * CH);
  _Float16* t256b  = allocH((size_t)2 * 256 * CH);
  _Float16* t128a  = allocH((size_t)2 * 128 * CH);
  _Float16* t128b  = allocH((size_t)2 * 128 * CH);
  _Float16* t64a   = allocH((size_t)2 * 64 * CH);
  _Float16* t64b   = allocH((size_t)2 * 64 * CH);
  _Float16* t64c   = allocH((size_t)2 * 64 * CH);
  _Float16* xh     = allocH((size_t)2 * 32 * CH);
  float* coordsb = allocF((size_t)2 * NPAD * 3);
  float* invs    = allocF((size_t)2 * NPAD);
  float* invl    = allocF((size_t)2 * NPAD);
  float* pooled  = allocF(1024);
  float* fc1     = allocF(512);
  float* fc2     = allocF(256);
  float* fc3     = allocF(8192);
  float* gmaxv   = allocF(1024);

  _Float16* xftm  = cat + (long)512 * CH;   // rows 512..575
  _Float16* xm2   = cat + (long)576 * CH;   // rows 576..1087
  _Float16* xglm2 = cat + (long)1088 * CH;  // rows 1088..1599
  const long catbs = (long)1600 * CH;

  // conv into f16 intermediate (full padded columns)
  auto conv = [&](const _Float16* x, long xbs, int wi, int bi, int bn, int relu,
                  int O, int I, _Float16* out, long obs, int row0) {
    dim3 grid(NPAD / 64, (O + 31) / 32, 2), block(128);
    conv_wmma_kernel<_Float16><<<grid, block, 0, stream>>>(
        x, xbs, F(wi), F(bi), BN(bn, 0), BN(bn, 1), BN(bn, 2), BN(bn, 3),
        relu, O, I, out, obs, NPAD, row0, NPAD);
  };

  // Adjacency prep + input conversion
  convert_x_kernel<<<(2 * 32 * NPAD + 255) / 256, 256, 0, stream>>>(F(0), xh);
  coords_copy_kernel<<<(2 * NPAD + 255) / 256, 256, 0, stream>>>(F(0), coordsb);
  rowsum_kernel<<<dim3(NPAD / 128, 2), 128, 0, stream>>>(coordsb, invs, invl);

  // MLP1
  conv(xh, 32 * CH, 1, 2, 3, 1, 64, 6, t64a, 64 * CH, 0);
  conv(t64a, 64 * CH, 7, 8, 9, 1, 64, 64, t64b, 64 * CH, 0);

  // STN (feature transform)
  conv(t64b, 64 * CH, 13, 14, 25, 1, 64, 64, t64a, 64 * CH, 0);
  conv(t64a, 64 * CH, 15, 16, 29, 1, 128, 64, t128a, 128 * CH, 0);
  conv(t128a, 128 * CH, 17, 18, 33, 1, 512, 128, t512a, 512 * CH, 0);
  maxpool_kernel<<<1024, 256, 0, stream>>>(t512a, 512 * CH, 512, NREAL, pooled);
  fc_kernel<<<(2 * 256 + 127) / 128, 128, 0, stream>>>(
      pooled, F(19), F(20), F(37), F(38), F(39), F(40), 1, 0, 512, 256, fc1);
  fc_kernel<<<(2 * 128 + 127) / 128, 128, 0, stream>>>(
      fc1, F(21), F(22), F(41), F(42), F(43), F(44), 1, 0, 256, 128, fc2);
  fc_kernel<<<(2 * 4096 + 127) / 128, 128, 0, stream>>>(
      fc2, F(23), F(24), nullptr, nullptr, nullptr, nullptr, 0, 1, 128, 4096,
      fc3);
  xftm_kernel<<<(2 * 64 * NPAD + 255) / 256, 256, 0, stream>>>(t64b, fc3, xftm,
                                                               catbs);

  // GLM1: sap(64ch) + two 32-ch branches + merge conv
  adj_gemm_kernel<0, 4><<<dim3(NPAD / 64, 1, 2), 128, 0, stream>>>(
      coordsb, invs, invl, xftm, catbs, t64a, 64 * CH, nullptr, 0);
  conv(xftm, catbs, 45, 46, 47, 1, 32, 64, t64c, 64 * CH, 0);
  conv(t64a, 64 * CH, 51, 52, 53, 1, 32, 64, t64c, 64 * CH, 32);
  conv(t64c, 64 * CH, 57, 58, 59, 1, 64, 64, t64a, 64 * CH, 0);

  // MLP2 (c3 writes directly into concat rows 576..1087)
  conv(t64a, 64 * CH, 63, 64, 65, 1, 64, 64, t64b, 64 * CH, 0);
  conv(t64b, 64 * CH, 69, 70, 71, 1, 128, 64, t128a, 128 * CH, 0);
  conv(t128a, 128 * CH, 75, 76, 77, 1, 512, 128, cat, catbs, 576);

  // Dual adjacency GEMM: sap1 (thr 0.1) and sap2 (thr 0.2) in one pass
  adj_gemm_kernel<1, 8><<<dim3(NPAD / 64, 4, 2), 128, 0, stream>>>(
      coordsb, invs, invl, xm2, catbs, t512a, 512 * CH, t512b, 512 * CH);

  // GLM2: three 512->128 branches into cat384, merge 384->512 into concat
  conv(xm2, catbs, 81, 82, 83, 1, 128, 512, cat384, 384 * CH, 0);
  conv(t512a, 512 * CH, 87, 88, 89, 1, 128, 512, cat384, 384 * CH, 128);
  conv(t512b, 512 * CH, 93, 94, 95, 1, 128, 512, cat384, 384 * CH, 256);
  conv(cat384, 384 * CH, 99, 100, 101, 1, 512, 384, cat, catbs, 1088);

  // Global max of x_glm2 broadcast into concat rows 0..511
  maxpool_kernel<<<1024, 256, 0, stream>>>(xglm2, catbs, 512, NREAL, gmaxv);
  broadcast_gmax_kernel<<<(2 * 512 * NPAD + 255) / 256, 256, 0, stream>>>(gmaxv,
                                                                          cat);

  // MLP3
  conv(cat, catbs, 105, 106, 107, 1, 256, 1600, t256a, 256 * CH, 0);
  conv(t256a, 256 * CH, 111, 112, 113, 1, 256, 256, t256b, 256 * CH, 0);
  conv(t256b, 256 * CH, 117, 118, 119, 1, 128, 256, t128a, 128 * CH, 0);
  conv(t128a, 128 * CH, 123, 124, 125, 1, 128, 128, t128b, 128 * CH, 0);

  // Heads: seg [2,17,6000] then edge [2,2,6000] as fp32 into d_out
  float* outp = (float*)d_out;
  {
    dim3 grid(NPAD / 64, 1, 2), block(128);
    conv_wmma_kernel<float><<<grid, block, 0, stream>>>(
        t128b, 128 * CH, F(129), F(130), nullptr, nullptr, nullptr, nullptr,
        0, 17, 128, outp, 17 * NREAL, NREAL, 0, NREAL);
    conv_wmma_kernel<float><<<grid, block, 0, stream>>>(
        t128b, 128 * CH, F(131), F(132), nullptr, nullptr, nullptr, nullptr,
        0, 2, 128, outp + (long)2 * 17 * NREAL, 2 * NREAL, NREAL, 0, NREAL);
  }
}
